// PointTransformerSeg_32169305047428
// MI455X (gfx1250) — matrix operand order repaired and checked by reading, not compiled, not run on hardware
//
#include <hip/hip_runtime.h>
#include <hip/hip_bf16.h>

typedef __attribute__((ext_vector_type(16))) _Float16 v16h;
typedef __attribute__((ext_vector_type(8)))  _Float16 v8h;
typedef __attribute__((ext_vector_type(8)))  float    v8f;

#define NPTS 16384

// ---------------------------------------------------------------------------
// WMMA GEMM: C[M,N] = epilogue( A[M,Kp](f16) @ W[N,Kp](f16)^T )
// epilogue: y = acc*scale[n] + bias[n]; (+resPre); relu?; (+resPost)
// One wave (32 threads) computes one 16x16 tile. Kp must be a multiple of 32.
// M must be a multiple of 16 (true for all call sites); N may be ragged.
// ---------------------------------------------------------------------------
__global__ __launch_bounds__(32) void wmma_gemm(
    const _Float16* __restrict__ Ap, int M, int Kp,
    const _Float16* __restrict__ Wp, int N,
    const float* __restrict__ scale, const float* __restrict__ bias,
    const float* __restrict__ resPre, const float* __restrict__ resPost,
    int relu, float* __restrict__ C) {
  const int lane   = threadIdx.x;        // 0..31
  const int laneHi = lane >> 4;          // 0 or 1
  const int l      = lane & 15;
  const int m0 = blockIdx.x * 16;
  const int n0 = blockIdx.y * 16;
  const int mA = m0 + l;                       // M multiple of 16 -> always valid
  const int nB = (n0 + l < N) ? (n0 + l) : (N - 1);   // clamp for ragged N

  v8f c = {};
  const _Float16* arow = Ap + (size_t)mA * Kp;
  const _Float16* brow = Wp + (size_t)nB * Kp;
  for (int k0 = 0; k0 < Kp; k0 += 32) {
    // A fragment (16-bit A 16x32 layout): lanes0-15 K groups {0-7,16-23},
    // lanes16-31 {8-15,24-31}
    const _Float16* ap = arow + k0 + laneHi * 8;
    v8h alo = *(const v8h*)(ap);
    v8h ahi = *(const v8h*)(ap + 16);
    v16h a;
#pragma unroll
    for (int i = 0; i < 8; ++i) { a[i] = alo[i]; a[8 + i] = ahi[i]; }
    // B fragment (16-bit B 32x16): same K order as A: lanes0-15 {0-7,16-23}, lanes16-31 {8-15,24-31}
    const _Float16* bp = brow + k0 + laneHi * 8;
    v8h blo = *(const v8h*)(bp);
    v8h bhi = *(const v8h*)(bp + 16);
    v16h b;
#pragma unroll
    for (int i = 0; i < 8; ++i) { b[i] = blo[i]; b[8 + i] = bhi[i]; }
    c = __builtin_amdgcn_wmma_f32_16x16x32_f16(
        /*neg_a=*/false, a, /*neg_b=*/false, b,
        /*c_mod=*/(short)0, c, /*reuse_a=*/false, /*reuse_b=*/false);
  }

  const int n = n0 + l;
  if (n < N) {
    const float s  = scale ? scale[n] : 1.0f;
    const float bb = bias ? bias[n] : 0.0f;
#pragma unroll
    for (int r = 0; r < 8; ++r) {
      const int m = m0 + r + laneHi * 8;
      if (m < M) {
        size_t off = (size_t)m * N + n;
        float y = c[r] * s + bb;
        if (resPre)  y += resPre[off];
        if (relu)    y = fmaxf(y, 0.0f);
        if (resPost) y += resPost[off];
        C[off] = y;
      }
    }
  }
}

// ---------------------------------------------------------------------------
// Packing kernels: f32 -> f16 with K padded to Kp (zeros)
// ---------------------------------------------------------------------------
__global__ void pack_pad(const float* __restrict__ src, int rows, int K, int Kp,
                         _Float16* __restrict__ dst) {
  size_t tid = (size_t)blockIdx.x * blockDim.x + threadIdx.x;
  size_t total = (size_t)rows * Kp;
  if (tid >= total) return;
  int kk = (int)(tid % Kp);
  size_t r = tid / Kp;
  dst[tid] = (_Float16)((kk < K) ? src[r * (size_t)K + kk] : 0.0f);
}

__global__ void pack_concat2(const float* __restrict__ a, int ca,
                             const float* __restrict__ b, int cb,
                             int rows, int Kp, _Float16* __restrict__ dst) {
  size_t tid = (size_t)blockIdx.x * blockDim.x + threadIdx.x;
  size_t total = (size_t)rows * Kp;
  if (tid >= total) return;
  int kk = (int)(tid % Kp);
  size_t r = tid / Kp;
  float v = 0.0f;
  if (kk < ca) v = a[r * ca + kk];
  else if (kk < ca + cb) v = b[r * cb + (kk - ca)];
  dst[tid] = (_Float16)v;
}

__global__ void pack_concat_bcast(const float* __restrict__ a, int ca,
                                  const float* __restrict__ bv, int cb,
                                  int rows, int Kp, _Float16* __restrict__ dst) {
  size_t tid = (size_t)blockIdx.x * blockDim.x + threadIdx.x;
  size_t total = (size_t)rows * Kp;
  if (tid >= total) return;
  int kk = (int)(tid % Kp);
  size_t r = tid / Kp;
  float v = 0.0f;
  if (kk < ca) v = a[r * ca + kk];
  else if (kk < ca + cb) v = bv[kk - ca];
  dst[tid] = (_Float16)v;
}

// grouped = concat(pos[nbr]-sub[q], x[nbr]) padded to Kp, rows = mq*16
__global__ void pack_grouped(const float* __restrict__ pos,
                             const float* __restrict__ x, int cfeat,
                             const float* __restrict__ sub,
                             const int* __restrict__ nbr,
                             int mq, int Kp, _Float16* __restrict__ dst) {
  size_t tid = (size_t)blockIdx.x * blockDim.x + threadIdx.x;
  size_t total = (size_t)mq * 16 * Kp;
  if (tid >= total) return;
  int kk = (int)(tid % Kp);
  size_t row = tid / Kp;
  int q = (int)(row >> 4);
  int j = nbr[row];
  float v = 0.0f;
  if (kk < 3) v = pos[(size_t)j * 3 + kk] - sub[(size_t)q * 3 + kk];
  else if (kk < 3 + cfeat) v = x[(size_t)j * cfeat + (kk - 3)];
  dst[tid] = (_Float16)v;
}

// ---------------------------------------------------------------------------
// scale/bias folding:  s = (g ? g*rsqrt(1+eps) : 1);  b = (linb ? s*linb : 0) + (be ? be : 0)
// ---------------------------------------------------------------------------
__global__ void make_affine(const float* __restrict__ g, const float* __restrict__ be,
                            const float* __restrict__ linb, int N,
                            float* __restrict__ s, float* __restrict__ b) {
  int i = blockIdx.x * blockDim.x + threadIdx.x;
  if (i >= N) return;
  const float inv = rsqrtf(1.0f + 1e-5f);
  float sv = g ? g[i] * inv : 1.0f;
  float bv = (linb ? sv * linb[i] : 0.0f) + (be ? be[i] : 0.0f);
  s[i] = sv;
  b[i] = bv;
}

// ---------------------------------------------------------------------------
// Farthest point sampling: single persistent block, LDS argmax tree.
// Matches reference: start at 0, argmax tie -> smallest index.
// ---------------------------------------------------------------------------
__global__ __launch_bounds__(1024) void fps_kernel(const float* __restrict__ pos,
                                                   int n, int m,
                                                   float* __restrict__ dist,
                                                   int* __restrict__ idx) {
  __shared__ float sv[1024];
  __shared__ int   si[1024];
  const int t = threadIdx.x;
  for (int i = t; i < n; i += 1024) dist[i] = 1e30f;
  if (t == 0) idx[0] = 0;
  __syncthreads();
  int last = 0;
  for (int it = 1; it < m; ++it) {
    const float lx = pos[(size_t)last * 3 + 0];
    const float ly = pos[(size_t)last * 3 + 1];
    const float lz = pos[(size_t)last * 3 + 2];
    float best = -1.0f;
    int bi = 0x7fffffff;
    for (int i = t; i < n; i += 1024) {
      float dx = pos[(size_t)i * 3 + 0] - lx;
      float dy = pos[(size_t)i * 3 + 1] - ly;
      float dz = pos[(size_t)i * 3 + 2] - lz;
      float d = dx * dx + dy * dy + dz * dz;
      float dd = fminf(dist[i], d);
      dist[i] = dd;
      if (dd > best || (dd == best && i < bi)) { best = dd; bi = i; }
    }
    sv[t] = best; si[t] = bi;
    __syncthreads();
    for (int s = 512; s > 0; s >>= 1) {
      if (t < s) {
        if (sv[t + s] > sv[t] || (sv[t + s] == sv[t] && si[t + s] < si[t])) {
          sv[t] = sv[t + s]; si[t] = si[t + s];
        }
      }
      __syncthreads();
    }
    last = si[0];
    if (t == 0) idx[it] = last;
    __syncthreads();
  }
}

__global__ void gather_rows3(const float* __restrict__ pos, const int* __restrict__ idx,
                             int m, float* __restrict__ sub) {
  int tid = blockIdx.x * blockDim.x + threadIdx.x;
  if (tid >= m * 3) return;
  int i = tid / 3, c = tid % 3;
  sub[tid] = pos[(size_t)idx[i] * 3 + c];
}

// ---------------------------------------------------------------------------
// Brute-force kNN (k=16): one thread per query
// ---------------------------------------------------------------------------
__global__ void knn16(const float* __restrict__ sub, int m,
                      const float* __restrict__ pos, int n,
                      int* __restrict__ nbr) {
  int q = blockIdx.x * blockDim.x + threadIdx.x;
  if (q >= m) return;
  const float qx = sub[(size_t)q * 3 + 0];
  const float qy = sub[(size_t)q * 3 + 1];
  const float qz = sub[(size_t)q * 3 + 2];
  float bd[16]; int bi[16];
#pragma unroll
  for (int t = 0; t < 16; ++t) { bd[t] = 3e38f; bi[t] = 0; }
  int wi = 0; float wd = 3e38f;
  for (int i = 0; i < n; ++i) {
    float dx = pos[(size_t)i * 3 + 0] - qx;
    float dy = pos[(size_t)i * 3 + 1] - qy;
    float dz = pos[(size_t)i * 3 + 2] - qz;
    float d = dx * dx + dy * dy + dz * dz;
    if (d < wd) {
      bd[wi] = d; bi[wi] = i;
      wd = bd[0]; wi = 0;
#pragma unroll
      for (int t = 1; t < 16; ++t) if (bd[t] > wd) { wd = bd[t]; wi = t; }
    }
  }
#pragma unroll
  for (int t = 0; t < 16; ++t) nbr[(size_t)q * 16 + t] = bi[t];
}

// kNN (k=3) with 1/d^2 normalized weights
__global__ void knn3(const float* __restrict__ ph, int nh,
                     const float* __restrict__ pl, int nl,
                     int* __restrict__ idx3, float* __restrict__ w3) {
  int i = blockIdx.x * blockDim.x + threadIdx.x;
  if (i >= nh) return;
  const float qx = ph[(size_t)i * 3 + 0];
  const float qy = ph[(size_t)i * 3 + 1];
  const float qz = ph[(size_t)i * 3 + 2];
  float d0 = 3e38f, d1 = 3e38f, d2 = 3e38f;
  int i0 = 0, i1 = 0, i2 = 0;
  for (int j = 0; j < nl; ++j) {
    float dx = pl[(size_t)j * 3 + 0] - qx;
    float dy = pl[(size_t)j * 3 + 1] - qy;
    float dz = pl[(size_t)j * 3 + 2] - qz;
    float d = dx * dx + dy * dy + dz * dz;
    if (d < d0)      { d2 = d1; i2 = i1; d1 = d0; i1 = i0; d0 = d; i0 = j; }
    else if (d < d1) { d2 = d1; i2 = i1; d1 = d; i1 = j; }
    else if (d < d2) { d2 = d; i2 = j; }
  }
  float w0 = 1.0f / fmaxf(d0, 1e-16f);
  float w1 = 1.0f / fmaxf(d1, 1e-16f);
  float w2 = 1.0f / fmaxf(d2, 1e-16f);
  float ws = w0 + w1 + w2;
  idx3[(size_t)i * 3 + 0] = i0; idx3[(size_t)i * 3 + 1] = i1; idx3[(size_t)i * 3 + 2] = i2;
  w3[(size_t)i * 3 + 0] = w0 / ws; w3[(size_t)i * 3 + 1] = w1 / ws; w3[(size_t)i * 3 + 2] = w2 / ws;
}

__global__ void interp_apply(const float* __restrict__ xs, int C,
                             const int* __restrict__ idx3, const float* __restrict__ w3,
                             size_t nh, float* __restrict__ out) {
  size_t tid = (size_t)blockIdx.x * blockDim.x + threadIdx.x;
  if (tid >= nh * C) return;
  size_t i = tid / C;
  int ch = (int)(tid % C);
  out[tid] = w3[i * 3 + 0] * xs[(size_t)idx3[i * 3 + 0] * C + ch]
           + w3[i * 3 + 1] * xs[(size_t)idx3[i * 3 + 1] * C + ch]
           + w3[i * 3 + 2] * xs[(size_t)idx3[i * 3 + 2] * C + ch];
}

__global__ void maxpool16(const float* __restrict__ h, int m, int C,
                          float* __restrict__ out) {
  size_t tid = (size_t)blockIdx.x * blockDim.x + threadIdx.x;
  if (tid >= (size_t)m * C) return;
  size_t q = tid / C;
  int ch = (int)(tid % C);
  float v = -3e38f;
#pragma unroll
  for (int r = 0; r < 16; ++r)
    v = fmaxf(v, h[((size_t)q * 16 + r) * C + ch]);
  out[tid] = v;
}

__global__ void colmean(const float* __restrict__ x, int rows, int C,
                        float* __restrict__ mean) {
  int ch = blockIdx.x * blockDim.x + threadIdx.x;
  if (ch >= C) return;
  float acc = 0.0f;
  for (int r = 0; r < rows; ++r) acc += x[(size_t)r * C + ch];
  mean[ch] = acc / (float)rows;
}

__global__ void gemv_relu(const float* __restrict__ W, const float* __restrict__ b,
                          const float* __restrict__ v, int N, int K,
                          float* __restrict__ out) {
  int o = blockIdx.x * blockDim.x + threadIdx.x;
  if (o >= N) return;
  float acc = b[o];
  for (int k = 0; k < K; ++k) acc += W[(size_t)o * K + k] * v[k];
  out[o] = fmaxf(acc, 0.0f);
}

// ---------------------------------------------------------------------------
// Host orchestration
// ---------------------------------------------------------------------------
extern "C" void kernel_launch(void* const* d_in, const int* in_sizes, int n_in,
                              void* d_out, int out_size, void* d_ws, size_t ws_size,
                              hipStream_t stream) {
  auto in = [&](int i) { return (const float*)d_in[i]; };
  const float* pos  = in(0);
  const float* feat = in(1);

  // ---- workspace bump allocator ----
  char* base = (char*)d_ws;
  size_t off = 0;
  auto alloc = [&](size_t bytes) -> void* {
    off = (off + 255) & ~(size_t)255;
    void* p = base + off;
    off += bytes;
    return p;
  };
  float* x1 = (float*)alloc((size_t)NPTS * 32 * 4);
  float* x2 = (float*)alloc((size_t)4096 * 64 * 4);
  float* x3 = (float*)alloc((size_t)1024 * 128 * 4);
  float* x4 = (float*)alloc((size_t)256 * 256 * 4);
  float* x5 = (float*)alloc((size_t)64 * 512 * 4);
  float* p2 = (float*)alloc((size_t)4096 * 3 * 4);
  float* p3 = (float*)alloc((size_t)1024 * 3 * 4);
  float* p4 = (float*)alloc((size_t)256 * 3 * 4);
  float* p5 = (float*)alloc((size_t)64 * 3 * 4);
  _Float16* PACKA = (_Float16*)alloc((size_t)65536 * 64 * 2);   // 8 MB max (enc2 grouped)
  _Float16* PACKB = (_Float16*)alloc((size_t)512 * 1024 * 2);   // 1 MB max (dec5 W1)
  float* HB     = (float*)alloc((size_t)65536 * 64 * 4);        // 16 MB max (enc2 pre-pool)
  float* XSB    = (float*)alloc((size_t)4096 * 64 * 4);
  float* INTERP = (float*)alloc((size_t)NPTS * 32 * 4);
  float* DIST   = (float*)alloc((size_t)NPTS * 4);
  int*   FIDX   = (int*)alloc((size_t)4096 * 4);
  int*   NBR    = (int*)alloc((size_t)65536 * 4);
  int*   IDX3   = (int*)alloc((size_t)NPTS * 3 * 4);
  float* W3     = (float*)alloc((size_t)NPTS * 3 * 4);
  float* MEANV  = (float*)alloc(512 * 4);
  float* GMEAN  = (float*)alloc(512 * 4);
  float* SC     = (float*)alloc(1024 * 4);
  float* BI     = (float*)alloc(1024 * 4);

  // ---- helpers ----
  auto gemm = [&](const _Float16* Ap, int M, int Kp, const _Float16* Wp, int N,
                  const float* s, const float* b, const float* pre,
                  const float* post, int relu, float* C) {
    dim3 g((M + 15) / 16, (N + 15) / 16);
    wmma_gemm<<<g, 32, 0, stream>>>(Ap, M, Kp, Wp, N, s, b, pre, post, relu, C);
  };
  auto pack = [&](const float* src, int rows, int K, int Kp, _Float16* dst) {
    size_t tot = (size_t)rows * Kp;
    pack_pad<<<(unsigned)((tot + 255) / 256), 256, 0, stream>>>(src, rows, K, Kp, dst);
  };
  auto affine = [&](const float* g, const float* be, const float* linb, int N) {
    make_affine<<<(N + 255) / 256, 256, 0, stream>>>(g, be, linb, N, SC, BI);
  };
  auto resblock = [&](float* x, int M, int C, int pb) {
    // blk: W1, g1, b1, W2, g2, b2
    affine(in(pb + 1), in(pb + 2), nullptr, C);
    pack(in(pb + 0), C, C, C, PACKB);
    pack(x, M, C, C, PACKA);
    gemm(PACKA, M, C, PACKB, C, SC, BI, nullptr, nullptr, 1, HB);
    affine(in(pb + 4), in(pb + 5), nullptr, C);
    pack(in(pb + 3), C, C, C, PACKB);
    pack(HB, M, C, C, PACKA);
    gemm(PACKA, M, C, PACKB, C, SC, BI, /*pre=*/x, nullptr, 1, x);  // relu(x + bn(h@W2))
  };
  auto tdown = [&](const float* posP, int n, const float* xP, int c,
                   const float* W, const float* g, const float* b,
                   int m, int cout, float* psub, float* xout) {
    const int K = 3 + c;
    const int Kp = ((K + 31) / 32) * 32;
    fps_kernel<<<1, 1024, 0, stream>>>(posP, n, m, DIST, FIDX);
    gather_rows3<<<(m * 3 + 255) / 256, 256, 0, stream>>>(posP, FIDX, m, psub);
    knn16<<<(m + 255) / 256, 256, 0, stream>>>(psub, m, posP, n, NBR);
    size_t tot = (size_t)m * 16 * Kp;
    pack_grouped<<<(unsigned)((tot + 255) / 256), 256, 0, stream>>>(
        posP, xP, c, psub, NBR, m, Kp, PACKA);
    pack(W, cout, K, Kp, PACKB);
    affine(g, b, nullptr, cout);
    gemm(PACKA, m * 16, Kp, PACKB, cout, SC, BI, nullptr, nullptr, 1, HB);
    size_t mt = (size_t)m * cout;
    maxpool16<<<(unsigned)((mt + 255) / 256), 256, 0, stream>>>(HB, m, cout, xout);
  };
  auto decstage = [&](const float* posH, int nh, float* xH,
                      const float* posL, int nl, const float* xL,
                      int cin, int cout, int bi) {
    // tu: W2,b2,g2,be2, W1,b1,g1,be1, blk(6)
    affine(in(bi + 2), in(bi + 3), in(bi + 1), cout);
    pack(xL, nl, cin, cin, PACKA);
    pack(in(bi + 0), cout, cin, cin, PACKB);
    gemm(PACKA, nl, cin, PACKB, cout, SC, BI, nullptr, nullptr, 1, XSB);
    knn3<<<(nh + 255) / 256, 256, 0, stream>>>(posH, nh, posL, nl, IDX3, W3);
    size_t it = (size_t)nh * cout;
    interp_apply<<<(unsigned)((it + 255) / 256), 256, 0, stream>>>(
        XSB, cout, IDX3, W3, (size_t)nh, INTERP);
    affine(in(bi + 6), in(bi + 7), in(bi + 5), cout);
    pack(xH, nh, cout, cout, PACKA);
    pack(in(bi + 4), cout, cout, cout, PACKB);
    gemm(PACKA, nh, cout, PACKB, cout, SC, BI, nullptr, /*post=*/INTERP, 1, xH);
    resblock(xH, nh, cout, bi + 8);
  };

  // ---- encoder ----
  // enc1: x1 = relu(bn(concat(pos,feat) @ W^T))   [16384, 32]
  {
    size_t tot = (size_t)NPTS * 32;
    pack_concat2<<<(unsigned)((tot + 255) / 256), 256, 0, stream>>>(
        pos, 3, feat, 3, NPTS, 32, PACKA);
    pack(in(2), 32, 6, 32, PACKB);
    affine(in(3), in(4), nullptr, 32);
    gemm(PACKA, NPTS, 32, PACKB, 32, SC, BI, nullptr, nullptr, 1, x1);
  }
  tdown(pos, NPTS, x1, 32, in(5),  in(6),  in(7),  4096, 64,  p2, x2);
  tdown(p2, 4096,  x2, 64, in(8),  in(9),  in(10), 1024, 128, p3, x3);
  tdown(p3, 1024,  x3, 128, in(11), in(12), in(13), 256, 256, p4, x4);
  tdown(p4, 256,   x4, 256, in(14), in(15), in(16), 64,  512, p5, x5);

  // ---- dec5 head ----
  // dec5: 17 W2, 18 b2, 19 W1, 20 b1, 21 g1, 22 be1, blk 23..28
  {
    colmean<<<(512 + 255) / 256, 256, 0, stream>>>(x5, 64, 512, MEANV);
    gemv_relu<<<(512 + 255) / 256, 256, 0, stream>>>(in(17), in(18), MEANV, 512, 512, GMEAN);
    size_t tot = (size_t)64 * 1024;
    pack_concat_bcast<<<(unsigned)((tot + 255) / 256), 256, 0, stream>>>(
        x5, 512, GMEAN, 512, 64, 1024, PACKA);
    pack(in(19), 512, 1024, 1024, PACKB);
    affine(in(21), in(22), in(20), 512);
    gemm(PACKA, 64, 1024, PACKB, 512, SC, BI, nullptr, nullptr, 1, x5);
    resblock(x5, 64, 512, 23);
  }

  // ---- decoder ladder ----
  decstage(p4, 256,  x4, p5, 64,   x5, 512, 256, 29);  // dec4
  decstage(p3, 1024, x3, p4, 256,  x4, 256, 128, 43);  // dec3
  decstage(p2, 4096, x2, p3, 1024, x3, 128, 64,  57);  // dec2
  decstage(pos, NPTS, x1, p2, 4096, x2, 64,  32,  71); // dec1

  // ---- classifier ----
  // cls: 85 W1, 86 b1, 87 g, 88 be, 89 W2, 90 b2
  {
    affine(in(87), in(88), in(86), 32);
    pack(x1, NPTS, 32, 32, PACKA);
    pack(in(85), 32, 32, 32, PACKB);
    gemm(PACKA, NPTS, 32, PACKB, 32, SC, BI, nullptr, nullptr, 1, HB);
    affine(nullptr, nullptr, in(90), 13);
    pack(HB, NPTS, 32, 32, PACKA);
    pack(in(89), 13, 32, 32, PACKB);
    gemm(PACKA, NPTS, 32, PACKB, 13, SC, BI, nullptr, nullptr, 0, (float*)d_out);
  }
}
